// MemoryNetwork_34239479283717
// MI455X (gfx1250) — matrix operand order repaired and checked by reading, not compiled, not run on hardware
//
#include <hip/hip_runtime.h>
#include <hip/hip_bf16.h>
#include <math.h>

#define DDIM   64
#define K3     192
#define TLEN   256
#define NROWS  4096     // BATCH * NUM_NODES
#define NWAVES 4        // waves per block
#define NBLOCKS 64      // 64 blocks * 4 waves = 256 waves = NROWS/16 tiles
#define SSTRIDE 72      // padded staging stride (bank-conflict free, 16B aligned rows)

typedef __bf16 bf16;
typedef __attribute__((ext_vector_type(16))) bf16  v16bf;
typedef __attribute__((ext_vector_type(8)))  bf16  bf16x8;
typedef __attribute__((ext_vector_type(8)))  float v8f;
typedef __attribute__((ext_vector_type(4)))  float f32x4;

// ws (bf16 element offsets)
#define TE_OFF   0        // [256][64] time encoding (mask-gathered)
#define WMSG_OFF 16384    // msg_W   [64][192] row-major == B^T for GEMM1
#define WIH_OFF  28672    // W_ih    [192][64] row-major == B^T for GEMM2
#define WHH_OFF  40960    // W_hh    [192][64] row-major == B^T for GEMM3

static __device__ __forceinline__ v8f wmma_bf16(v16bf a, v16bf b, v8f c) {
  return __builtin_amdgcn_wmma_f32_16x16x32_bf16(false, a, false, b, (short)0, c, false, false);
}

static __device__ __forceinline__ v16bf ldpair(const bf16* p0, const bf16* p1) {
  bf16x8 lo = *reinterpret_cast<const bf16x8*>(p0);   // 16B load
  bf16x8 hi = *reinterpret_cast<const bf16x8*>(p1);   // 16B load
  v16bf r;
#pragma unroll
  for (int i = 0; i < 8; ++i) { r[i] = lo[i]; r[8 + i] = hi[i]; }
  return r;
}

// B fragment (K-slice x 16 cols). WT[n][k] row-major, B[k][n] = WT[n][k].
// ISA: K = (r&3)*2 + (r>>2)*16 + 8*half => two contiguous 8-elem runs (matches A).
static __device__ __forceinline__ v16bf bfrag(const bf16* __restrict__ WT, int stride,
                                              int n0, int kb, int lane) {
  const int half = lane >> 4;
  const int nn   = (lane & 15) + n0;
  const bf16* p = WT + nn * stride + kb + 8 * half;
  return ldpair(p, p + 16);
}

// A fragment (16x32) from bf16 staging [16][SSTRIDE].
// ISA: VGPR r -> K = (r&3)*2 + (r>>2)*16 + 8*half  => two contiguous 8-elem runs.
static __device__ __forceinline__ v16bf afrag_s(const bf16* __restrict__ S, int kb, int lane) {
  const int half = lane >> 4;
  const int m    = lane & 15;
  const bf16* p = S + m * SSTRIDE + kb + 8 * half;
  return ldpair(p, p + 16);
}

// A fragment from f32 global rows (row stride DDIM), converted to bf16.
static __device__ __forceinline__ v16bf afrag_x(const float* __restrict__ xrow, int kb, int lane) {
  const int half = lane >> 4;
  const int m    = lane & 15;
  const float* p = xrow + m * DDIM + kb + 8 * half;
  f32x4 a0 = *reinterpret_cast<const f32x4*>(p);
  f32x4 a1 = *reinterpret_cast<const f32x4*>(p + 4);
  f32x4 b0 = *reinterpret_cast<const f32x4*>(p + 16);
  f32x4 b1 = *reinterpret_cast<const f32x4*>(p + 20);
  v16bf r;
#pragma unroll
  for (int i = 0; i < 4; ++i) {
    r[i]      = (bf16)a0[i];
    r[4 + i]  = (bf16)a1[i];
    r[8 + i]  = (bf16)b0[i];
    r[12 + i] = (bf16)b1[i];
  }
  return r;
}

// A fragment from bf16 time-encoding row (same values for every M row).
static __device__ __forceinline__ v16bf afrag_t(const bf16* __restrict__ te, int kb, int lane) {
  const int half = lane >> 4;
  const bf16* p = te + kb + 8 * half;
  return ldpair(p, p + 16);
}

// ---- fast activations: use gfx1250 hardware v_tanh_f32 when available ----
static __device__ __forceinline__ float fast_rcp(float v) {
#if __has_builtin(__builtin_amdgcn_rcpf)
  return __builtin_amdgcn_rcpf(v);
#else
  return 1.0f / v;
#endif
}

static __device__ __forceinline__ float fast_tanh(float v) {
#if __has_builtin(__builtin_amdgcn_tanhf)
  return __builtin_amdgcn_tanhf(v);
#elif __has_builtin(__builtin_amdgcn_tanh_f32)
  return __builtin_amdgcn_tanh_f32(v);
#else
  float vc = fminf(fmaxf(v, -18.0f), 18.0f);   // avoid exp overflow; tanh saturated anyway
  float e  = __expf(-2.0f * vc);
  return (1.0f - e) * fast_rcp(1.0f + e);
#endif
}

static __device__ __forceinline__ float sigm(float v) {
  return fmaf(0.5f, fast_tanh(0.5f * v), 0.5f);
}

// tanh-approx GELU (error ~1e-3, below bf16 matmul noise)
static __device__ __forceinline__ float gelu_f(float v) {
  float v2 = v * v;
  float u  = v * fmaf(0.0356774081f, v2, 0.7978845608f);
  return 0.5f * v * (1.0f + fast_tanh(u));
}

// -------- prep: bake te (mask-gathered) + bf16 weights into workspace --------
__global__ void memnet_prep_34239479283717(const int* __restrict__ mask,
                                           const float* __restrict__ freq,
                                           const float* __restrict__ phase,
                                           const float* __restrict__ msg_W,
                                           const float* __restrict__ W_ih,
                                           const float* __restrict__ W_hh,
                                           bf16* __restrict__ ws) {
  int i = blockIdx.x * blockDim.x + threadIdx.x;
  if (i < TLEN * DDIM) {
    int t = i / DDIM, d = i % DDIM;
    float tg = (float)(t * mask[t]);                    // tr[t] = t * mask[t]
    ws[TE_OFF + i] = (bf16)cosf(tg * freq[d] + phase[d]);
  }
  if (i < DDIM * K3) {
    ws[WMSG_OFF + i] = (bf16)msg_W[i];
    ws[WIH_OFF + i]  = (bf16)W_ih[i];
    ws[WHH_OFF + i]  = (bf16)W_hh[i];
  }
}

// -------- main: one wave owns a 16-row tile for the full T=256 scan --------
__global__ __launch_bounds__(NWAVES * 32) void
memnet_main_34239479283717(const float* __restrict__ x,
                           const int* __restrict__ mask,
                           const float* __restrict__ msg_b,
                           const float* __restrict__ b_ih,
                           const float* __restrict__ b_hh,
                           const bf16* __restrict__ ws,
                           float* __restrict__ out) {
  __shared__ bf16 stage[NWAVES][16 * SSTRIDE];   // per-wave s/msg relayout buffer

  const int tid  = threadIdx.x;
  const int lane = tid & 31;
  const int wave = tid >> 5;
  const int half = lane >> 4;
  const int nn   = lane & 15;

  const bf16* te   = ws + TE_OFF;
  const bf16* Wmsg = ws + WMSG_OFF;   // [64][192]
  const bf16* Wih  = ws + WIH_OFF;    // [192][64]
  const bf16* Whh  = ws + WHH_OFF;    // [192][64]

  const int tile = blockIdx.x * NWAVES + wave;   // 0..255
  const int R0   = tile * 16;
  bf16* ss = stage[wave];

  // biases in C/D layout (per-lane column nn+16j, same for all 8 rows)
  float bias_msg[4], bias_rz[8], bias_in[4], bias_hn[4];
#pragma unroll
  for (int j = 0; j < 4; ++j) bias_msg[j] = msg_b[16 * j + nn];
#pragma unroll
  for (int j = 0; j < 8; ++j) bias_rz[j] = b_ih[16 * j + nn] + b_hh[16 * j + nn];
#pragma unroll
  for (int j = 0; j < 4; ++j) {
    bias_in[j] = b_ih[16 * (8 + j) + nn];
    bias_hn[j] = b_hh[16 * (8 + j) + nn];
  }

  // ---- s0 = mean_t x[t, tile rows] : fully-coalesced linear accumulation ----
  f32x4 acc[8];
#pragma unroll
  for (int q = 0; q < 8; ++q) acc[q] = (f32x4)0.0f;
  const float* xb = x + (size_t)R0 * DDIM + (size_t)lane * 32;
  for (int t = 0; t < TLEN; ++t) {
    const float* p = xb + (size_t)t * NROWS * DDIM;
#pragma unroll
    for (int q = 0; q < 8; ++q) acc[q] += *reinterpret_cast<const f32x4*>(p + q * 4);
  }
  {
    const float inv = 1.0f / (float)TLEN;
    const int m  = lane >> 1;
    const int c0 = (lane & 1) * 32;
#pragma unroll
    for (int q = 0; q < 8; ++q)
#pragma unroll
      for (int i = 0; i < 4; ++i)
        ss[m * SSTRIDE + c0 + q * 4 + i] = (bf16)(acc[q][i] * inv);
  }
  __builtin_amdgcn_wave_barrier();

  // f32 state in C/D fragment layout: sfrag[j][e] = s[e+8*half][16j+nn]
  v8f sfrag[4];
#pragma unroll
  for (int j = 0; j < 4; ++j)
#pragma unroll
    for (int e = 0; e < 8; ++e)
      sfrag[j][e] = (float)ss[(e + 8 * half) * SSTRIDE + 16 * j + nn];

  // ---- time scan ----
  for (int t = 0; t < TLEN; ++t) {
    const int tg = t * mask[t];
    const float* xt  = x + (size_t)tg * NROWS * DDIM + (size_t)R0 * DDIM;
    const bf16*  tet = te + t * DDIM;

    if (t + 1 < TLEN) {  // prefetch next timestep's x tile
      const int tg2 = (t + 1) * mask[t + 1];
      __builtin_prefetch(x + (size_t)tg2 * NROWS * DDIM + (size_t)R0 * DDIM + lane * 32, 0, 0);
    }

    v16bf xA0 = afrag_x(xt, 0, lane),  xA1 = afrag_x(xt, 32, lane);
    v16bf sA0 = afrag_s(ss, 0, lane),  sA1 = afrag_s(ss, 32, lane);
    v16bf tA0 = afrag_t(tet, 0, lane), tA1 = afrag_t(tet, 32, lane);
    __builtin_amdgcn_wave_barrier();   // sA reads complete before ss reuse below

    // GEMM1: msg = gelu([x | s | te] @ msg_W^T + msg_b), K=192, N=64
#pragma unroll
    for (int j = 0; j < 4; ++j) {
      v8f c;
#pragma unroll
      for (int e = 0; e < 8; ++e) c[e] = bias_msg[j];
      c = wmma_bf16(xA0, bfrag(Wmsg, K3, 16 * j,   0, lane), c);
      c = wmma_bf16(xA1, bfrag(Wmsg, K3, 16 * j,  32, lane), c);
      c = wmma_bf16(sA0, bfrag(Wmsg, K3, 16 * j,  64, lane), c);
      c = wmma_bf16(sA1, bfrag(Wmsg, K3, 16 * j,  96, lane), c);
      c = wmma_bf16(tA0, bfrag(Wmsg, K3, 16 * j, 128, lane), c);
      c = wmma_bf16(tA1, bfrag(Wmsg, K3, 16 * j, 160, lane), c);
#pragma unroll
      for (int e = 0; e < 8; ++e) {
        float v = gelu_f(c[e]);
        ss[(e + 8 * half) * SSTRIDE + 16 * j + nn] = (bf16)v;   // stage msg for A relayout
      }
    }
    __builtin_amdgcn_wave_barrier();
    v16bf mA0 = afrag_s(ss, 0, lane), mA1 = afrag_s(ss, 32, lane);

    // GEMM2+3 fused with GRU elementwise, one 16-column slab at a time
    // (keeps only 4 live accumulators -> low VGPR pressure)
    float* ot = out + (size_t)t * NROWS * DDIM + (size_t)R0 * DDIM;
#pragma unroll
    for (int j = 0; j < 4; ++j) {
      v8f cr, cz, ci, ch;
#pragma unroll
      for (int e = 0; e < 8; ++e) {
        cr[e] = bias_rz[j];
        cz[e] = bias_rz[4 + j];
        ci[e] = bias_in[j];
        ch[e] = bias_hn[j];
      }
      // r gate: cols 16j..16j+15 of gi+gh
      cr = wmma_bf16(mA0, bfrag(Wih, DDIM, 16 * j,        0, lane), cr);
      cr = wmma_bf16(mA1, bfrag(Wih, DDIM, 16 * j,       32, lane), cr);
      cr = wmma_bf16(sA0, bfrag(Whh, DDIM, 16 * j,        0, lane), cr);
      cr = wmma_bf16(sA1, bfrag(Whh, DDIM, 16 * j,       32, lane), cr);
      // z gate: cols 64+16j..
      cz = wmma_bf16(mA0, bfrag(Wih, DDIM, 16 * (4 + j),  0, lane), cz);
      cz = wmma_bf16(mA1, bfrag(Wih, DDIM, 16 * (4 + j), 32, lane), cz);
      cz = wmma_bf16(sA0, bfrag(Whh, DDIM, 16 * (4 + j),  0, lane), cz);
      cz = wmma_bf16(sA1, bfrag(Whh, DDIM, 16 * (4 + j), 32, lane), cz);
      // n gate inputs: cols 128+16j..
      ci = wmma_bf16(mA0, bfrag(Wih, DDIM, 16 * (8 + j),  0, lane), ci);
      ci = wmma_bf16(mA1, bfrag(Wih, DDIM, 16 * (8 + j), 32, lane), ci);
      ch = wmma_bf16(sA0, bfrag(Whh, DDIM, 16 * (8 + j),  0, lane), ch);
      ch = wmma_bf16(sA1, bfrag(Whh, DDIM, 16 * (8 + j), 32, lane), ch);

#pragma unroll
      for (int e = 0; e < 8; ++e) {
        float r  = sigm(cr[e]);
        float z  = sigm(cz[e]);
        float nv = fast_tanh(fmaf(r, ch[e], ci[e]));
        float sn = fmaf(z, sfrag[j][e] - nv, nv);   // (1-z)*n + z*s
        sfrag[j][e] = sn;
        const int row = e + 8 * half;
        const int col = 16 * j + nn;
        ot[row * DDIM + col] = sn;
        ss[row * SSTRIDE + col] = (bf16)sn;
      }
    }
    __builtin_amdgcn_wave_barrier();
  }
}

extern "C" void kernel_launch(void* const* d_in, const int* in_sizes, int n_in,
                              void* d_out, int out_size, void* d_ws, size_t ws_size,
                              hipStream_t stream) {
  (void)in_sizes; (void)n_in; (void)out_size; (void)ws_size;
  const float* x      = (const float*)d_in[0];
  const int*   mask   = (const int*)d_in[1];
  const float* msg_W  = (const float*)d_in[2];
  const float* msg_b  = (const float*)d_in[3];
  const float* W_ih   = (const float*)d_in[4];
  const float* W_hh   = (const float*)d_in[5];
  const float* b_ih   = (const float*)d_in[6];
  const float* b_hh   = (const float*)d_in[7];
  const float* freq   = (const float*)d_in[8];
  const float* phase  = (const float*)d_in[9];
  float* out = (float*)d_out;
  bf16*  ws  = (bf16*)d_ws;

  memnet_prep_34239479283717<<<(TLEN * DDIM + 255) / 256, 256, 0, stream>>>(
      mask, freq, phase, msg_W, W_ih, W_hh, ws);
  memnet_main_34239479283717<<<NBLOCKS, NWAVES * 32, 0, stream>>>(
      x, mask, msg_b, b_ih, b_hh, ws, out);
}
